// MultiHeadAttention_16518444220980
// MI455X (gfx1250) — compile-verified, matrix operand order checked by reading, not run on hardware
//
#include <hip/hip_runtime.h>

// ---------------------------------------------------------------------------
// Hydra attention block for MI455X (gfx1250, wave32, WMMA).
// bf16 tensor-core GEMMs (v_wmma_f32_16x16x32_bf16) with double-buffered
// async global->LDS staging (global_load_async_to_lds_b128 / ASYNCcnt).
// Weights are converted f32->bf16 AND pre-transposed to [N][K] so both A and
// B tiles stage as contiguous 16B chunks along K.
// ---------------------------------------------------------------------------

typedef unsigned short u16;
typedef __attribute__((ext_vector_type(16))) __bf16 v16bf;
typedef __attribute__((ext_vector_type(8)))  float  v8f;
typedef __attribute__((address_space(3))) u16 lds_u16;

#define B_ 4
#define S_ 2048
#define E_ 1024
#define H_ 8
#define D_ 1024
#define MROWS (B_ * S_)   // 8192

__device__ __forceinline__ u16 f2bf(float x) {
  union { float f; unsigned u; } c; c.f = x;
  unsigned r = c.u + 0x7FFFu + ((c.u >> 16) & 1u);   // round-to-nearest-even
  return (u16)(r >> 16);
}
__device__ __forceinline__ float bf2f(u16 h) {
  union { float f; unsigned u; } c; c.u = ((unsigned)h) << 16;
  return c.f;
}

// 16B async copy global -> LDS, tracked by ASYNCcnt (CDNA5).
__device__ __forceinline__ void async_b128(u16* lds, const u16* g) {
  asm volatile("global_load_async_to_lds_b128 %0, %1, off"
               :: "v"((lds_u16*)lds), "v"(g) : "memory");
}
__device__ __forceinline__ void wait_async_4() {
  asm volatile("s_wait_asynccnt 0x4" ::: "memory");
}
__device__ __forceinline__ void wait_async_0() {
  asm volatile("s_wait_asynccnt 0x0" ::: "memory");
}

// ---------------------------------------------------------------------------
// f32 -> bf16 bulk conversion (vectorized, grid-stride)
// ---------------------------------------------------------------------------
__global__ void cvt_f32_bf16(const float* __restrict__ src, u16* __restrict__ dst,
                             long n4) {
  long i = (long)blockIdx.x * blockDim.x + threadIdx.x;
  long stride = (long)gridDim.x * blockDim.x;
  const float4* s4 = (const float4*)src;
  uint2* d2 = (uint2*)dst;
  for (long j = i; j < n4; j += stride) {
    float4 v = s4[j];
    union { u16 h[4]; uint2 u; } o;
    o.h[0] = f2bf(v.x); o.h[1] = f2bf(v.y); o.h[2] = f2bf(v.z); o.h[3] = f2bf(v.w);
    d2[j] = o.u;
  }
}

// ---------------------------------------------------------------------------
// f32 [R][C] -> bf16 transposed [C][R], per-head, 32x32 LDS tiles
// ---------------------------------------------------------------------------
__global__ __launch_bounds__(256)
void cvtT_f32_bf16(const float* __restrict__ src, u16* __restrict__ dst,
                   int R, int C, long sStrideH, long dStrideH) {
  __shared__ float tile[32][33];
  int h = blockIdx.z;
  src += (long)h * sStrideH;
  dst += (long)h * dStrideH;
  int r0 = blockIdx.y * 32, c0 = blockIdx.x * 32;
  int tx = threadIdx.x & 31;
  int ty = threadIdx.x >> 5;           // 0..7
  #pragma unroll
  for (int i = 0; i < 4; ++i)
    tile[ty + i * 8][tx] = src[(long)(r0 + ty + i * 8) * C + c0 + tx];
  __syncthreads();
  #pragma unroll
  for (int i = 0; i < 4; ++i)
    dst[(long)(c0 + ty + i * 8) * R + r0 + tx] = f2bf(tile[tx][ty + i * 8]);
}

__global__ void zero_f32(float* __restrict__ p, int n) {
  int i = blockIdx.x * blockDim.x + threadIdx.x;
  if (i < n) p[i] = 0.f;
}

// ---------------------------------------------------------------------------
// Batched bf16 GEMM, f32 accumulate, C[h] = A[h] (MxK) * Bt[h]^T + bias[h]
//   A  : row-major [M][K]      (lda = K stride)
//   Bt : row-major [N][K]      (pre-transposed weights, ldb = K stride)
// Tile 128x128, BK=32, 8 wave32 waves (4x2), 2x4 WMMA frags per wave.
// Double-buffered LDS filled with global_load_async_to_lds_b128.
// ---------------------------------------------------------------------------
template<bool OUT_BF16>
__global__ __launch_bounds__(256)
void gemm_bf16_wmma(const u16* __restrict__ A, const u16* __restrict__ Bt,
                    const float* __restrict__ bias, void* __restrict__ Cv,
                    int K, int lda, int ldb, int ldc,
                    long aStrideH, long bStrideH, long cStrideH, int biasStrideH)
{
  __shared__ u16 As[2][128][40];   // +8 pad: 80B row stride (16B aligned, conflict-free)
  __shared__ u16 Bs[2][128][40];

  const int h = blockIdx.z;
  const long m0 = (long)blockIdx.y * 128;
  const long n0 = (long)blockIdx.x * 128;
  const u16* Abase = A  + (long)h * aStrideH + m0 * (long)lda;
  const u16* Bbase = Bt + (long)h * bStrideH + n0 * (long)ldb;
  const float* bp = bias ? (bias + (long)h * biasStrideH) : nullptr;

  const int t    = threadIdx.x;
  const int wave = t >> 5;
  const int lane = t & 31;
  const int hlf  = lane >> 4;
  const int idx  = lane & 15;
  const int wm   = wave >> 1;      // 0..3 -> m offset wm*32
  const int wn   = wave & 1;       // 0..1 -> n offset wn*64

  // per-thread staging chunks: 512 chunks of 16B per tile, 2 per thread
  const int r0c = t >> 2,        c0c = (t & 3) << 3;          // chunk 0
  const int r1c = (t + 256) >> 2, c1c = ((t + 256) & 3) << 3; // chunk 1

  v8f acc[2][4] = {};
  union Frag { v16bf v; uint4 u[2]; };

  const int ntiles = K >> 5;

  auto issue = [&](int kt, int buf) {
    const long ko = (long)(kt << 5);
    async_b128(&As[buf][r0c][c0c], Abase + (long)r0c * lda + ko + c0c);
    async_b128(&Bs[buf][r0c][c0c], Bbase + (long)r0c * ldb + ko + c0c);
    async_b128(&As[buf][r1c][c1c], Abase + (long)r1c * lda + ko + c1c);
    async_b128(&Bs[buf][r1c][c1c], Bbase + (long)r1c * ldb + ko + c1c);
  };

  issue(0, 0);

  for (int it = 0; it < ntiles; ++it) {
    const int cur = it & 1;
    const bool more = (it + 1) < ntiles;
    if (more) issue(it + 1, cur ^ 1);     // prefetch next tile into other buffer
    if (more) wait_async_4();             // tile `it`'s 4 copies done (in-order)
    else      wait_async_0();
    __syncthreads();                      // all waves' copies visible

    Frag af[2], bfr[4];
    #pragma unroll
    for (int i = 0; i < 2; ++i) {
      int r = wm * 32 + i * 16 + idx;
      af[i].u[0] = *(const uint4*)&As[cur][r][hlf * 8];
      af[i].u[1] = *(const uint4*)&As[cur][r][hlf * 8 + 16];
    }
    #pragma unroll
    for (int j = 0; j < 4; ++j) {
      int r = wn * 64 + j * 16 + idx;
      bfr[j].u[0] = *(const uint4*)&Bs[cur][r][hlf * 8];
      bfr[j].u[1] = *(const uint4*)&Bs[cur][r][hlf * 8 + 16];
    }
    #pragma unroll
    for (int i = 0; i < 2; ++i)
      #pragma unroll
      for (int j = 0; j < 4; ++j)
        acc[i][j] = __builtin_amdgcn_wmma_f32_16x16x32_bf16(
            false, af[i].v, false, bfr[j].v, (short)0, acc[i][j], false, false);
    __syncthreads();                      // done reading `cur` before it is refilled
  }

  // epilogue: D elem v <-> m = v + hlf*8, n = idx
  u16*   Cb = (u16*)Cv   + (long)h * cStrideH;
  float* Cf = (float*)Cv + (long)h * cStrideH;
  #pragma unroll
  for (int i = 0; i < 2; ++i) {
    #pragma unroll
    for (int j = 0; j < 4; ++j) {
      long gn = n0 + wn * 64 + j * 16 + idx;
      float bv = bp ? bp[gn] : 0.f;
      #pragma unroll
      for (int v = 0; v < 8; ++v) {
        long gm = m0 + wm * 32 + i * 16 + hlf * 8 + v;
        float r = acc[i][j][v] + bv;
        if (OUT_BF16) Cb[gm * (long)ldc + gn] = f2bf(r);
        else          Cf[gm * (long)ldc + gn] = r;
      }
    }
  }
}

// ---------------------------------------------------------------------------
// pooled[h,b,e] = sum_s mask? 0 : (k[h,b,s,e]/||k row||) * v[h,b,s,e]
// qkv layout: [H][B*S][3E] bf16; one wave per row, 4 rows per wave.
// ---------------------------------------------------------------------------
__global__ __launch_bounds__(256)
void pool_kv(const u16* __restrict__ qkv, const unsigned char* __restrict__ mask,
             float* __restrict__ pooled) {
  __shared__ float pl[E_];
  const int t = threadIdx.x;
  for (int i = t; i < E_; i += 256) pl[i] = 0.f;
  __syncthreads();

  const int chunks = S_ / 32;
  int chunk = blockIdx.x % chunks;
  int hb    = blockIdx.x / chunks;
  int h = hb / B_, b = hb % B_;
  int wave = t >> 5, lane = t & 31;

  for (int r = 0; r < 4; ++r) {
    int s = chunk * 32 + wave * 4 + r;
    size_t row = ((size_t)h * MROWS + (size_t)b * S_ + s) * (3 * E_);
    const u16* kp = qkv + row + E_;
    const u16* vp = qkv + row + 2 * E_;
    float ss = 0.f;
    for (int e = lane; e < E_; e += 32) { float kv = bf2f(kp[e]); ss += kv * kv; }
    #pragma unroll
    for (int off = 16; off > 0; off >>= 1) ss += __shfl_xor(ss, off, 32);
    float coef = (mask[b * S_ + s] != 0) ? 0.f : rsqrtf(ss + 1e-30f);
    for (int e = lane; e < E_; e += 32)
      atomicAdd(&pl[e], coef * bf2f(kp[e]) * bf2f(vp[e]));
  }
  __syncthreads();
  float* gp = pooled + ((size_t)h * B_ + b) * E_;
  for (int i = t; i < E_; i += 256) atomicAdd(&gp[i], pl[i]);
}

// ---------------------------------------------------------------------------
// ah[h, b*S+s, e] = bf16( pooled[h,b,e] * q[h,b,s,e] / ||q row|| )
// ---------------------------------------------------------------------------
__global__ __launch_bounds__(256)
void q_scale(const u16* __restrict__ qkv, const float* __restrict__ pooled,
             u16* __restrict__ ah) {
  int t = threadIdx.x, wave = t >> 5, lane = t & 31;
  long gid = (long)blockIdx.x * 8 + wave;
  int h   = (int)(gid / MROWS);
  int rem = (int)(gid % MROWS);                    // b*S + s
  const u16*   qp = qkv + ((size_t)h * MROWS + rem) * (3 * E_);
  const float* pp = pooled + ((size_t)h * B_ + (rem / S_)) * E_;
  u16*         ap = ah + ((size_t)h * MROWS + rem) * E_;
  float ss = 0.f;
  for (int e = lane; e < E_; e += 32) { float q = bf2f(qp[e]); ss += q * q; }
  #pragma unroll
  for (int off = 16; off > 0; off >>= 1) ss += __shfl_xor(ss, off, 32);
  float rn = rsqrtf(ss + 1e-30f);
  for (int e = lane; e < E_; e += 32)
    ap[e] = f2bf(bf2f(qp[e]) * rn * pp[e]);
}

// ---------------------------------------------------------------------------
// launcher
// ---------------------------------------------------------------------------
extern "C" void kernel_launch(void* const* d_in, const int* in_sizes, int n_in,
                              void* d_out, int out_size, void* d_ws, size_t ws_size,
                              hipStream_t stream) {
  (void)in_sizes; (void)n_in; (void)out_size; (void)ws_size;
  const float* x    = (const float*)d_in[0];   // [B,S,E]
  const float* Wqkv = (const float*)d_in[1];   // [H,E,3E]
  const float* bqkv = (const float*)d_in[2];   // [H,3E]
  const float* Wh   = (const float*)d_in[3];   // [H,E,D]
  const float* bh   = (const float*)d_in[4];   // [H,D]
  const float* Wo   = (const float*)d_in[5];   // [H*D,D]
  const float* bo   = (const float*)d_in[6];   // [D]
  const unsigned char* mask = (const unsigned char*)d_in[7];  // [B,S] bool
  float* out = (float*)d_out;                  // [B,S,D] f32

  char* ws = (char*)d_ws;
  u16* xb       = (u16*)ws;  ws += (size_t)MROWS * E_ * 2;            // 16 MB
  u16* Wqkv_t   = (u16*)ws;  ws += (size_t)H_ * 3 * E_ * E_ * 2;      // 48 MB  [H][3E][E]
  u16* Wh_t     = (u16*)ws;  ws += (size_t)H_ * D_ * E_ * 2;          // 16 MB  [H][D][E]
  u16* Wo_t     = (u16*)ws;  ws += (size_t)D_ * H_ * D_ * 2;          // 16 MB  [D][H*D]
  u16* qkv      = (u16*)ws;  ws += (size_t)H_ * MROWS * 3 * E_ * 2;   // 384 MB [H][B*S][3E]
  float* pooled = (float*)ws; ws += (size_t)H_ * B_ * E_ * 4;         // 128 KB
  u16* ah       = (u16*)ws;  ws += (size_t)H_ * MROWS * E_ * 2;       // 128 MB [H][B*S][E]
  u16* cat      = (u16*)ws;  ws += (size_t)MROWS * H_ * D_ * 2;       // 128 MB [B*S][H*D]

  // x: straight convert (stays row-major [M][K])
  {
    long n4 = (long)MROWS * E_ / 4;
    long grid = (n4 + 255) / 256; if (grid > 4096) grid = 4096;
    cvt_f32_bf16<<<dim3((unsigned)grid), 256, 0, stream>>>(x, xb, n4);
  }
  // weights: convert + transpose to [N][K]
  cvtT_f32_bf16<<<dim3(3 * E_ / 32, E_ / 32, H_), 256, 0, stream>>>(
      Wqkv, Wqkv_t, E_, 3 * E_, (long)E_ * 3 * E_, (long)3 * E_ * E_);
  cvtT_f32_bf16<<<dim3(D_ / 32, E_ / 32, H_), 256, 0, stream>>>(
      Wh, Wh_t, E_, D_, (long)E_ * D_, (long)D_ * E_);
  cvtT_f32_bf16<<<dim3(D_ / 32, (H_ * D_) / 32, 1), 256, 0, stream>>>(
      Wo, Wo_t, H_ * D_, D_, 0L, 0L);

  zero_f32<<<(H_ * B_ * E_ + 255) / 256, 256, 0, stream>>>(pooled, H_ * B_ * E_);

  // GEMM1: qkv[h] = xb * Wqkv_t[h]^T + bqkv[h]     [8192 x 3072], K=1024
  gemm_bf16_wmma<true><<<dim3(3 * E_ / 128, MROWS / 128, H_), 256, 0, stream>>>(
      xb, Wqkv_t, bqkv, qkv,
      /*K*/E_, /*lda*/E_, /*ldb*/E_, /*ldc*/3 * E_,
      /*aStrideH*/0L, /*bStrideH*/(long)3 * E_ * E_,
      /*cStrideH*/(long)MROWS * 3 * E_, /*biasStrideH*/3 * E_);

  pool_kv<<<H_ * B_ * (S_ / 32), 256, 0, stream>>>(qkv, mask, pooled);
  q_scale<<<(H_ * MROWS) / 8, 256, 0, stream>>>(qkv, pooled, ah);

  // GEMM2: cat[:, h*D:(h+1)*D] = ah[h] * Wh_t[h]^T + bh[h]   [8192 x 1024], K=1024
  gemm_bf16_wmma<true><<<dim3(D_ / 128, MROWS / 128, H_), 256, 0, stream>>>(
      ah, Wh_t, bh, cat,
      /*K*/E_, /*lda*/E_, /*ldb*/E_, /*ldc*/H_ * D_,
      /*aStrideH*/(long)MROWS * E_, /*bStrideH*/(long)D_ * E_,
      /*cStrideH*/(long)D_, /*biasStrideH*/D_);

  // GEMM3: out = cat * Wo_t^T + bo                [8192 x 1024], K=8192
  gemm_bf16_wmma<false><<<dim3(D_ / 128, MROWS / 128, 1), 256, 0, stream>>>(
      cat, Wo_t, bo, out,
      /*K*/H_ * D_, /*lda*/H_ * D_, /*ldb*/H_ * D_, /*ldc*/D_,
      0L, 0L, 0L, 0);
}
